// HATSModel_5231270167366
// MI455X (gfx1250) — matrix operand order repaired and checked by reading, not compiled, not run on hardware
//
#include <hip/hip_runtime.h>
#include <stdint.h>
#include <stddef.h>

// ---------------- problem constants ----------------
#define N_BATCH 8192
#define T_STEPS 64
#define DF      6
#define HID     128
#define G3      384        // 3*HID
#define EPSV    1e-5f

// ---------------- workspace layout (bytes) ----------------
#define OFF_WB    0u          // 3 bf16 weight matrices [384*128] : 3*98304 = 294912
#define OFF_H2    294912u     // final hidden h2 [8192*128] f32 : 4194304
#define OFF_M     4489216u    // row means m[8192] f32 : 32768
#define OFF_MU1   4521984u    // 128 f32
#define OFF_RS1   4522496u    // 128 f32
#define OFF_G     4523008u    // Gram [128*128] f32 : 65536
#define OFF_P     4588544u    // p[128] f32 : 512
#define OFF_FT    4589056u    // F^T bf16 [128*128] : 32768
#define OFF_Q     4621824u    // q[128] f32 : 512
#define OFF_OFC   4622336u    // out_fc [8192*128] f32 : 4194304
#define OFF_MU2   8816640u
#define OFF_RS2   8817152u

typedef __attribute__((ext_vector_type(16))) __bf16   v16bf;
typedef __attribute__((ext_vector_type(8)))  float    v8f;
typedef __attribute__((ext_vector_type(4)))  unsigned u32x4;
typedef __attribute__((ext_vector_type(8)))  int      i32x8;
typedef __attribute__((ext_vector_type(4)))  int      i32x4;

// ---------------- small helpers ----------------
__device__ __forceinline__ unsigned short f2bf(float f) {
  unsigned u = __builtin_bit_cast(unsigned, f);
  u += 0x7fffu + ((u >> 16) & 1u);          // round-to-nearest-even
  return (unsigned short)(u >> 16);
}
__device__ __forceinline__ float bflo(unsigned u) { return __builtin_bit_cast(float, u << 16); }
__device__ __forceinline__ float bfhi(unsigned u) { return __builtin_bit_cast(float, u & 0xffff0000u); }
__device__ __forceinline__ float sigm(float x)    { return 1.0f / (1.0f + __expf(-x)); }

union VB { u32x4 u[2]; v16bf v; };

// A fragment (16x32 bf16) per ISA layout: lanes 0-15 rows M=0..15 hold K {0..7,16..23},
// lanes 16-31 same rows hold K {8..15,24..31}. hbase = row-major [16][128] bf16 in LDS.
__device__ __forceinline__ v16bf load_afrag(const char* hbase, int lane, int kc) {
  int hi = lane >> 4, m = lane & 15;
  int kb = kc * 32 + hi * 8;
  const char* rp = hbase + m * 256 + kb * 2;
  VB t; t.u[0] = *(const u32x4*)rp; t.u[1] = *(const u32x4*)(rp + 32);
  return t.v;
}
// B fragment (32x16 bf16): lane holds one column n = lane%16, K (lane<16 ? {0..7,16..23} : {8..15,24..31})
// (matches A). B[k,n] = W[gc+n, kbase+k]; W row-major [*][128] bf16, row stride 256 B.
__device__ __forceinline__ v16bf load_bfrag(const char* wbase, int lane, int gc, int kc) {
  int hi = lane >> 4, n = lane & 15;
  const char* rp = wbase + (gc + n) * 256 + (kc * 64 + hi * 16);
  VB t; t.u[0] = *(const u32x4*)rp; t.u[1] = *(const u32x4*)(rp + 32);
  return t.v;
}
__device__ __forceinline__ v8f wmma_bf16(v16bf a, v16bf b, v8f c) {
  return __builtin_amdgcn_wmma_f32_16x16x32_bf16(false, a, false, b, (short)0, c, false, false);
}

// ---------------- TDM: DMA a 2-D bf16 tile global -> LDS (CDNA5 Tensor Data Mover) ----
// D# packing per CDNA5 ISA ch.8: group0 = {count/type/lds_addr/global_addr},
// group1 = {data_size, tensor dims, tile dims, dim0 stride}. Full-tensor tile.
// This toolchain exposes the 6-arg builtin: (g0 v4u, g1 v8i, g2 v4i, g3 v4i, ext v8i, cpol).
#if __has_builtin(__builtin_amdgcn_tensor_load_to_lds)
#define HAVE_TDM 1
__device__ __forceinline__ void tdm_load_2d_bf16(unsigned lds_off, const void* gptr,
                                                 unsigned width, unsigned height,
                                                 unsigned stride) {
  unsigned long long ga = (unsigned long long)(uintptr_t)gptr;
  u32x4 g0 = { 1u,                                              // count=1, user descriptor
               lds_off,                                         // lds_addr [63:32]
               (unsigned)ga,                                    // global_addr lo
               ((unsigned)(ga >> 32) & 0x01ffffffu) | (2u << 30) };  // addr[56:32] | type=2
  i32x8 g1 = { (int)(1u << 16),                                 // data_size=1 -> 2 bytes/elem
               (int)(width << 16),                              // tensor_dim0[15:0] @ bits 63:48
               (int)(((height & 0xffffu) << 16) | (width >> 16)),// tensor_dim1 lo | dim0 hi
               (int)((width << 16) | (height >> 16)),           // tile_dim0 | tensor_dim1 hi
               (int)(height & 0xffffu),                         // tile_dim1 (tile_dim2 = 0)
               (int)stride,                                     // tensor_dim0_stride[31:0]
               0, 0 };
  i32x4 z4 = {0, 0, 0, 0};
  i32x8 z8 = {0, 0, 0, 0, 0, 0, 0, 0};
  __builtin_amdgcn_tensor_load_to_lds(g0, g1, z4, z4, z8, 0);
}
#else
#define HAVE_TDM 0
#endif

// ---------------- K0: f32 -> bf16 weight conversion ----------------
__global__ void k_cvt(const float* __restrict__ whh0, const float* __restrict__ wih1,
                      const float* __restrict__ whh1, unsigned short* __restrict__ dst) {
  int i = blockIdx.x * 256 + threadIdx.x;
  if (i >= 3 * G3 * HID) return;
  float v = (i < 49152) ? whh0[i] : (i < 98304 ? wih1[i - 49152] : whh1[i - 98304]);
  dst[i] = f2bf(v);
}

// ---------------- K1: fused 2-layer GRU, 16 batch rows per block ----------------
__global__ void __launch_bounds__(256) k_gru(
    const float* __restrict__ x,
    const float* __restrict__ wih0, const float* __restrict__ bih0, const float* __restrict__ bhh0,
    const float* __restrict__ bih1, const float* __restrict__ bhh1,
    const unsigned short* __restrict__ whh0b, const unsigned short* __restrict__ wih1b,
    const unsigned short* __restrict__ whh1b, float* __restrict__ h2out) {
  extern __shared__ char smem[];
  char* s_whh0 = smem;                 // 98304
  char* s_wih1 = smem + 98304;         // 98304
  char* s_whh1 = smem + 196608;        // 98304
  char* s_x    = smem + 294912;        // [64][16][8] bf16 = 16384
  char* s_h1   = smem + 311296;        // [16][128] bf16 = 4096
  char* s_h2   = smem + 315392;        // 4096  (total 319488 B)

  const int tid = threadIdx.x;
  const int n0  = blockIdx.x * 16;

#if HAVE_TDM
  // Async-tensor DMA of all three bf16 weight matrices into LDS (one wave issues;
  // TENSORcnt tracks completion; barrier publishes to the workgroup).
  if (tid < 32) {
    tdm_load_2d_bf16(0u,      whh0b, HID, G3, HID);
    tdm_load_2d_bf16(98304u,  wih1b, HID, G3, HID);
    tdm_load_2d_bf16(196608u, whh1b, HID, G3, HID);
    __builtin_amdgcn_s_wait_tensorcnt(0);
  }
#else
  { // weights -> LDS (16B vectors)
    const u32x4* g0 = (const u32x4*)whh0b;
    const u32x4* g1 = (const u32x4*)wih1b;
    const u32x4* g2 = (const u32x4*)whh1b;
    u32x4* d0 = (u32x4*)s_whh0; u32x4* d1 = (u32x4*)s_wih1; u32x4* d2 = (u32x4*)s_whh1;
    for (int i = tid; i < 6144; i += 256) { d0[i] = g0[i]; d1[i] = g1[i]; d2[i] = g2[i]; }
  }
#endif
  // x tile transposed to time-major bf16: s_x[t][r][d], d padded to 8
  for (int i = tid; i < 8192; i += 256) {
    int d = i & 7, r = (i >> 3) & 15, t = i >> 7;
    float v = (d < DF) ? x[(size_t)(n0 + r) * 384 + d * 64 + t] : 0.0f;
    ((unsigned short*)s_x)[i] = f2bf(v);
  }
  for (int i = tid; i < 2048; i += 256) { ((unsigned*)s_h1)[i] = 0u; ((unsigned*)s_h2)[i] = 0u; }
  __syncthreads();

  const int w = tid >> 5, lane = tid & 31, hi = lane >> 4, nn = lane & 15;
  const int cr = w * 16 + nn;                 // hidden column slice of this wave/lane
  const int gr = w * 16, gz = 128 + w * 16, gn = 256 + w * 16;

  const float bh0r = bhh0[cr], bh0z = bhh0[128 + cr], bh0n = bhh0[256 + cr];
  const float bi1r = bih1[cr], bi1z = bih1[128 + cr], bi1n = bih1[256 + cr];
  const float bh1r = bhh1[cr], bh1z = bhh1[128 + cr], bh1n = bhh1[256 + cr];
  const float bi0r = bih0[cr], bi0z = bih0[128 + cr], bi0n = bih0[256 + cr];
  float wr[6], wz[6], wn[6];
  #pragma unroll
  for (int d = 0; d < 6; ++d) {
    wr[d] = wih0[cr * 6 + d];
    wz[d] = wih0[(128 + cr) * 6 + d];
    wn[d] = wih0[(256 + cr) * 6 + d];
  }

  float h1o[8], h2o[8];
  #pragma unroll
  for (int v = 0; v < 8; ++v) { h1o[v] = 0.f; h2o[v] = 0.f; }

  for (int t = 0; t < T_STEPS; ++t) {
    // Phase A: read A fragments of h1(t-1), h2(t-1)
    v16bf a1[4], a2[4];
    #pragma unroll
    for (int kc = 0; kc < 4; ++kc) { a1[kc] = load_afrag(s_h1, lane, kc); a2[kc] = load_afrag(s_h2, lane, kc); }
    __syncthreads();

    // gh0 = h1 @ whh0^T + bhh0 ; gh1 = h2 @ whh1^T + bhh1 (old h2)
    v8f g0r = {bh0r,bh0r,bh0r,bh0r,bh0r,bh0r,bh0r,bh0r};
    v8f g0z = {bh0z,bh0z,bh0z,bh0z,bh0z,bh0z,bh0z,bh0z};
    v8f g0n = {bh0n,bh0n,bh0n,bh0n,bh0n,bh0n,bh0n,bh0n};
    v8f g1r = {bh1r,bh1r,bh1r,bh1r,bh1r,bh1r,bh1r,bh1r};
    v8f g1z = {bh1z,bh1z,bh1z,bh1z,bh1z,bh1z,bh1z,bh1z};
    v8f g1n = {bh1n,bh1n,bh1n,bh1n,bh1n,bh1n,bh1n,bh1n};
    #pragma unroll
    for (int kc = 0; kc < 4; ++kc) {
      g0r = wmma_bf16(a1[kc], load_bfrag(s_whh0, lane, gr, kc), g0r);
      g0z = wmma_bf16(a1[kc], load_bfrag(s_whh0, lane, gz, kc), g0z);
      g0n = wmma_bf16(a1[kc], load_bfrag(s_whh0, lane, gn, kc), g0n);
      g1r = wmma_bf16(a2[kc], load_bfrag(s_whh1, lane, gr, kc), g1r);
      g1z = wmma_bf16(a2[kc], load_bfrag(s_whh1, lane, gz, kc), g1z);
      g1n = wmma_bf16(a2[kc], load_bfrag(s_whh1, lane, gn, kc), g1n);
    }

    // gi0 (K=6, VALU) + layer-0 gate combine -> h1(t)
    #pragma unroll
    for (int v = 0; v < 8; ++v) {
      int row = v + 8 * hi;
      u32x4 xv = *(const u32x4*)(s_x + ((t * 16 + row) << 4));
      float x0 = bflo(xv.x), x1 = bfhi(xv.x), x2 = bflo(xv.y);
      float x3 = bfhi(xv.y), x4 = bflo(xv.z), x5 = bfhi(xv.z);
      float ir = bi0r + x0*wr[0] + x1*wr[1] + x2*wr[2] + x3*wr[3] + x4*wr[4] + x5*wr[5];
      float iz = bi0z + x0*wz[0] + x1*wz[1] + x2*wz[2] + x3*wz[3] + x4*wz[4] + x5*wz[5];
      float in_ = bi0n + x0*wn[0] + x1*wn[1] + x2*wn[2] + x3*wn[3] + x4*wn[4] + x5*wn[5];
      float r  = sigm(ir + g0r[v]);
      float z  = sigm(iz + g0z[v]);
      float nh = tanhf(in_ + r * g0n[v]);
      float hn = (1.f - z) * nh + z * h1o[v];
      h1o[v] = hn;
      ((unsigned short*)s_h1)[row * 128 + cr] = f2bf(hn);
    }
    __syncthreads();

    // gi1 = h1(t) @ wih1^T + bih1 ; layer-1 combine -> h2(t)
    v8f i1r = {bi1r,bi1r,bi1r,bi1r,bi1r,bi1r,bi1r,bi1r};
    v8f i1z = {bi1z,bi1z,bi1z,bi1z,bi1z,bi1z,bi1z,bi1z};
    v8f i1n = {bi1n,bi1n,bi1n,bi1n,bi1n,bi1n,bi1n,bi1n};
    #pragma unroll
    for (int kc = 0; kc < 4; ++kc) {
      v16bf a1n = load_afrag(s_h1, lane, kc);
      i1r = wmma_bf16(a1n, load_bfrag(s_wih1, lane, gr, kc), i1r);
      i1z = wmma_bf16(a1n, load_bfrag(s_wih1, lane, gz, kc), i1z);
      i1n = wmma_bf16(a1n, load_bfrag(s_wih1, lane, gn, kc), i1n);
    }
    #pragma unroll
    for (int v = 0; v < 8; ++v) {
      int row = v + 8 * hi;
      float r  = sigm(i1r[v] + g1r[v]);
      float z  = sigm(i1z[v] + g1z[v]);
      float nh = tanhf(i1n[v] + r * g1n[v]);
      float hn = (1.f - z) * nh + z * h2o[v];
      h2o[v] = hn;
      ((unsigned short*)s_h2)[row * 128 + cr] = f2bf(hn);
    }
    __syncthreads();
  }

  #pragma unroll
  for (int v = 0; v < 8; ++v) {
    int row = v + 8 * hi;
    h2out[(size_t)(n0 + row) * 128 + cr] = h2o[v];
  }
}

// ---------------- K2/K6: per-column batch-norm stats ----------------
__global__ void __launch_bounds__(256) k_colstats(const float* __restrict__ src,
                                                  float* __restrict__ mu, float* __restrict__ rs) {
  __shared__ float red[512];
  const int c = blockIdx.x, tid = threadIdx.x;
  float s = 0.f, s2 = 0.f;
  for (int i = tid; i < N_BATCH; i += 256) {
    float v = src[(size_t)i * 128 + c];
    s += v; s2 += v * v;
  }
  red[tid] = s; red[256 + tid] = s2;
  __syncthreads();
  for (int off = 128; off; off >>= 1) {
    if (tid < off) { red[tid] += red[tid + off]; red[256 + tid] += red[256 + tid + off]; }
    __syncthreads();
  }
  if (tid == 0) {
    float m = red[0] * (1.0f / N_BATCH);
    float var = red[256] * (1.0f / N_BATCH) - m * m;
    mu[c] = m; rs[c] = rsqrtf(var + EPSV);
  }
}

// ---------------- K3a: per-row mean of bn1(h2) ----------------
__global__ void k_rowmean(const float* __restrict__ h2, const float* __restrict__ mu,
                          const float* __restrict__ rs, const float* __restrict__ g,
                          const float* __restrict__ b, float* __restrict__ m) {
  int n = blockIdx.x * 256 + threadIdx.x;
  if (n >= N_BATCH) return;
  float s = 0.f;
  for (int c = 0; c < 128; ++c)
    s += (h2[(size_t)n * 128 + c] - mu[c]) * rs[c] * g[c] + b[c];
  m[n] = s * (1.0f / 128.0f);
}

// ---------------- K3b: Gram G = hid^T hid and p = hid^T m (atomic partials) ----------------
__global__ void __launch_bounds__(256) k_gram(const float* __restrict__ h2, const float* __restrict__ mu,
                                              const float* __restrict__ rs, const float* __restrict__ g,
                                              const float* __restrict__ b, const float* __restrict__ m,
                                              float* __restrict__ G, float* __restrict__ p) {
  extern __shared__ char sm[];
  float* rows = (float*)sm;            // [128][128]
  float* mrow = (float*)(sm + 65536);  // [128]
  const int tid = threadIdx.x, r0 = blockIdx.x * 128;
  for (int i = tid; i < 16384; i += 256) {
    int r = i >> 7, c = i & 127;
    rows[i] = (h2[(size_t)(r0 + r) * 128 + c] - mu[c]) * rs[c] * g[c] + b[c];
  }
  if (tid < 128) mrow[tid] = m[r0 + tid];
  __syncthreads();
  for (int i = tid; i < 16384; i += 256) {
    int a = i >> 7, c = i & 127;
    float acc = 0.f;
    for (int r = 0; r < 128; ++r) acc += rows[r * 128 + a] * rows[r * 128 + c];
    atomicAdd(&G[i], acc);
  }
  if (tid < 128) {
    float acc = 0.f;
    for (int r = 0; r < 128; ++r) acc += mrow[r] * rows[r * 128 + tid];
    atomicAdd(&p[tid], acc);
  }
}

// ---------------- K4: F^T = (G @ fc_w^T / H)^T (bf16), q = fc_w @ p ----------------
__global__ void __launch_bounds__(256) k_ffold(const float* __restrict__ G, const float* __restrict__ p,
                                               const float* __restrict__ fcw,
                                               unsigned short* __restrict__ Ft, float* __restrict__ q) {
  const int tid = threadIdx.x;
  for (int i = tid; i < 16384; i += 256) {
    int j = i >> 7, c = i & 127;
    float acc = 0.f;
    for (int k = 0; k < 128; ++k) acc += G[j * 128 + k] * fcw[c * 128 + k];
    Ft[c * 128 + j] = f2bf(acc * (1.0f / 128.0f));   // Ft[c][j] = F[j][c]
  }
  if (tid < 128) {
    float acc = 0.f;
    for (int j = 0; j < 128; ++j) acc += fcw[tid * 128 + j] * p[j];
    q[tid] = acc;
  }
}

// ---------------- K5: out_fc = bn1(h2) @ F - m q^T + fc_b  (WMMA) ----------------
__global__ void __launch_bounds__(256) k_outfc(const float* __restrict__ h2, const float* __restrict__ mu,
                                               const float* __restrict__ rs, const float* __restrict__ g,
                                               const float* __restrict__ b, const float* __restrict__ m,
                                               const unsigned short* __restrict__ Ft, const float* __restrict__ q,
                                               const float* __restrict__ fcb, float* __restrict__ ofc) {
  extern __shared__ char sm[];
  char*  sA   = sm;                       // [16][128] bf16 = 4096
  char*  sF   = sm + 4096;                // [128][128] bf16 = 32768
  float* sm16 = (float*)(sm + 4096 + 32768);
  const int tid = threadIdx.x, n0 = blockIdx.x * 16;

#if HAVE_TDM
  if (tid < 32) {
    tdm_load_2d_bf16(4096u, Ft, HID, HID, HID);   // F^T tile via Tensor Data Mover
    __builtin_amdgcn_s_wait_tensorcnt(0);
  }
#else
  { const unsigned* src = (const unsigned*)Ft; unsigned* dst = (unsigned*)sF;
    for (int i = tid; i < 8192; i += 256) dst[i] = src[i]; }
#endif
  for (int i = tid; i < 2048; i += 256) {
    int r = i >> 7, c = i & 127;
    float v = (h2[(size_t)(n0 + r) * 128 + c] - mu[c]) * rs[c] * g[c] + b[c];
    ((unsigned short*)sA)[i] = f2bf(v);
  }
  if (tid < 16) sm16[tid] = m[n0 + tid];
  __syncthreads();

  const int w = tid >> 5, lane = tid & 31, hi = lane >> 4, nn = lane & 15;
  const int c = w * 16 + nn;
  const float fb = fcb[c], qc = q[c];
  v8f acc = {fb, fb, fb, fb, fb, fb, fb, fb};
  #pragma unroll
  for (int kc = 0; kc < 4; ++kc) {
    v16bf a  = load_afrag(sA, lane, kc);
    v16bf bb = load_bfrag(sF, lane, w * 16, kc);
    acc = wmma_bf16(a, bb, acc);
  }
  #pragma unroll
  for (int v = 0; v < 8; ++v) {
    int row = v + 8 * hi;
    ofc[(size_t)(n0 + row) * 128 + c] = acc[v] - sm16[row] * qc;
  }
}

// ---------------- K7: bn2 + leaky relu + output head ----------------
__global__ void k_final(const float* __restrict__ ofc, const float* __restrict__ mu,
                        const float* __restrict__ rs, const float* __restrict__ g,
                        const float* __restrict__ b, const float* __restrict__ fcow,
                        const float* __restrict__ fcob, float* __restrict__ out) {
  int n = blockIdx.x * 256 + threadIdx.x;
  if (n >= N_BATCH) return;
  float acc = fcob[0];
  for (int c = 0; c < 128; ++c) {
    float v = (ofc[(size_t)n * 128 + c] - mu[c]) * rs[c] * g[c] + b[c];
    v = v > 0.f ? v : 0.01f * v;
    acc += v * fcow[c];
  }
  out[n] = acc;
}

// ---------------- host ----------------
extern "C" void kernel_launch(void* const* d_in, const int* in_sizes, int n_in,
                              void* d_out, int out_size, void* d_ws, size_t ws_size,
                              hipStream_t stream) {
  const float* x    = (const float*)d_in[0];
  const float* wih0 = (const float*)d_in[1];
  const float* whh0 = (const float*)d_in[2];
  const float* bih0 = (const float*)d_in[3];
  const float* bhh0 = (const float*)d_in[4];
  const float* wih1 = (const float*)d_in[5];
  const float* whh1 = (const float*)d_in[6];
  const float* bih1 = (const float*)d_in[7];
  const float* bhh1 = (const float*)d_in[8];
  const float* bn1g = (const float*)d_in[9];
  const float* bn1b = (const float*)d_in[10];
  const float* fcw  = (const float*)d_in[11];
  const float* fcb  = (const float*)d_in[12];
  const float* bn2g = (const float*)d_in[13];
  const float* bn2b = (const float*)d_in[14];
  const float* fcow = (const float*)d_in[15];
  const float* fcob = (const float*)d_in[16];
  (void)in_sizes; (void)n_in; (void)out_size; (void)ws_size;

  char* ws = (char*)d_ws;
  unsigned short* wb    = (unsigned short*)(ws + OFF_WB);
  unsigned short* whh0b = wb;
  unsigned short* wih1b = wb + 49152;
  unsigned short* whh1b = wb + 98304;
  float* h2g = (float*)(ws + OFF_H2);
  float* mrw = (float*)(ws + OFF_M);
  float* mu1 = (float*)(ws + OFF_MU1);
  float* rs1 = (float*)(ws + OFF_RS1);
  float* Gb  = (float*)(ws + OFF_G);
  float* pb  = (float*)(ws + OFF_P);
  unsigned short* Ftb = (unsigned short*)(ws + OFF_FT);
  float* qb  = (float*)(ws + OFF_Q);
  float* ofc = (float*)(ws + OFF_OFC);
  float* mu2 = (float*)(ws + OFF_MU2);
  float* rs2 = (float*)(ws + OFF_RS2);

  k_cvt<<<576, 256, 0, stream>>>(whh0, wih1, whh1, wb);

  (void)hipFuncSetAttribute((const void*)k_gru, hipFuncAttributeMaxDynamicSharedMemorySize, 319488);
  k_gru<<<512, 256, 319488, stream>>>(x, wih0, bih0, bhh0, bih1, bhh1,
                                      whh0b, wih1b, whh1b, h2g);

  k_colstats<<<128, 256, 0, stream>>>(h2g, mu1, rs1);
  k_rowmean<<<32, 256, 0, stream>>>(h2g, mu1, rs1, bn1g, bn1b, mrw);

  (void)hipMemsetAsync(Gb, 0, 66048, stream);   // G (65536) + p (512), contiguous
  k_gram<<<64, 256, 66048, stream>>>(h2g, mu1, rs1, bn1g, bn1b, mrw, Gb, pb);
  k_ffold<<<1, 256, 0, stream>>>(Gb, pb, fcw, Ftb, qb);

  (void)hipFuncSetAttribute((const void*)k_outfc, hipFuncAttributeMaxDynamicSharedMemorySize, 36928);
  k_outfc<<<512, 256, 36928, stream>>>(h2g, mu1, rs1, bn1g, bn1b, mrw, Ftb, qb, fcb, ofc);

  k_colstats<<<128, 256, 0, stream>>>(ofc, mu2, rs2);
  k_final<<<32, 256, 0, stream>>>(ofc, mu2, rs2, bn2g, bn2b, fcow, fcob, (float*)d_out);
}
